// LocalAugmentation_24816321036988
// MI455X (gfx1250) — matrix operand order repaired and checked by reading, not compiled, not run on hardware
//
#include <hip/hip_runtime.h>

// ---------------- types ----------------
typedef __attribute__((ext_vector_type(16))) __bf16 bf16x16;
typedef __attribute__((ext_vector_type(8)))  __bf16 bf16x8;
typedef __attribute__((ext_vector_type(4)))  __bf16 bf16x4;
typedef __attribute__((ext_vector_type(8)))  float  f32x8;

#define HID   256
#define HEADS 8
#define ROWS_PER_BLOCK 64
#define BLOCK_THREADS  128        // 4 waves (wave32)
#define LDST  264                 // padded LDS row stride (bf16) -> 528B, conflict-free b128
#define ARR   (ROWS_PER_BLOCK * LDST)

__device__ __forceinline__ f32x8 vzero8() {
  f32x8 v;
#pragma unroll
  for (int i = 0; i < 8; ++i) v[i] = 0.0f;
  return v;
}

// A fragment (16x32 bf16, M x K): lane = row; lanes 0-15: K {0..7,16..23}, lanes 16-31: K {8..15,24..31}
__device__ __forceinline__ bf16x16 lds_afrag(const __bf16* base, int row, int kb, int hi) {
  const __bf16* p = base + row * LDST + kb * 32 + (hi ? 8 : 0);
  bf16x8 a = *(const bf16x8*)p;
  bf16x8 b = *(const bf16x8*)(p + 16);
  return __builtin_shufflevector(a, b, 0,1,2,3,4,5,6,7,8,9,10,11,12,13,14,15);
}

// B fragment (32x16 bf16, K x N): lane = column; lanes 0-15: K {0..7,16..23}, lanes 16-31: K {8..15,24..31}
// column n of W^T == row n of W (contiguous in memory)
__device__ __forceinline__ bf16x16 gbl_bfrag(const __bf16* w, int col, int kb, int hi) {
  const __bf16* p = w + col * HID + kb * 32 + hi * 8;
  bf16x8 a = *(const bf16x8*)p;
  bf16x8 b = *(const bf16x8*)(p + 16);
  return __builtin_shufflevector(a, b, 0,1,2,3,4,5,6,7,8,9,10,11,12,13,14,15);
}

#define WMMA(a, b, c) \
  __builtin_amdgcn_wmma_f32_16x16x32_bf16(false, (a), false, (b), (short)0, (c), false, false)

// ---------------- weight f32 -> bf16 conversion ----------------
__global__ void la_cvt_weights(const float* __restrict__ wq, const float* __restrict__ wk,
                               const float* __restrict__ wv, const float* __restrict__ wo,
                               __bf16* __restrict__ dst) {
  int t = blockIdx.x * blockDim.x + threadIdx.x;
  if (t < HID * HID) {
    dst[t]                 = (__bf16)wq[t];
    dst[HID * HID + t]     = (__bf16)wk[t];
    dst[2 * HID * HID + t] = (__bf16)wv[t];
    dst[3 * HID * HID + t] = (__bf16)wo[t];
  }
}

// ---------------- staging helper: f32 global -> bf16 LDS ----------------
__device__ __forceinline__ void stage_tile(const float* __restrict__ src, __bf16* dst,
                                           int row0, int tid) {
  // 64 rows x 256 cols, as float4 (64 f4 per row), 4096 f4 total
  for (int i = tid; i < ROWS_PER_BLOCK * (HID / 4); i += BLOCK_THREADS) {
    int r  = i >> 6;
    int c4 = i & 63;
    float4 v = ((const float4*)(src + (size_t)(row0 + r) * HID))[c4];
    bf16x4 hv;
    hv[0] = (__bf16)v.x; hv[1] = (__bf16)v.y; hv[2] = (__bf16)v.z; hv[3] = (__bf16)v.w;
    *(bf16x4*)(dst + r * LDST + c4 * 4) = hv;
  }
}

// ---------------- fused attention kernel ----------------
// LDS (132 KB) limits us to 2 blocks/WGP = 2 waves/SIMD; cap VGPRs accordingly (512).
__global__ void __launch_bounds__(BLOCK_THREADS, 2)
la_fused(const float* __restrict__ fine, const float* __restrict__ coarse,
         const float* __restrict__ motif, const __bf16* __restrict__ wbf,
         const float* __restrict__ bo, float* __restrict__ out) {
  extern __shared__ __align__(16) char smem_raw[];
  __bf16* sact = (__bf16*)smem_raw;      // [3][64][264]  fine, coarse, motif (bf16)
  __bf16* sctx = sact + 3 * ARR;         // [64][264]     per-wave context rows

  const __bf16* wq = wbf;
  const __bf16* wk = wbf + HID * HID;
  const __bf16* wv = wbf + 2 * HID * HID;
  const __bf16* wo = wbf + 3 * HID * HID;

  const int tid  = threadIdx.x;
  const int row0 = blockIdx.x * ROWS_PER_BLOCK;

  // warm L0/L2 with the head-0 weight slices (one 128B line per thread per matrix)
  __builtin_prefetch((const void*)(wq + (size_t)tid * 64), 0, 1);
  __builtin_prefetch((const void*)(wk + (size_t)tid * 64), 0, 1);
  __builtin_prefetch((const void*)(wv + (size_t)tid * 64), 0, 1);

  stage_tile(fine,   sact,            row0, tid);
  stage_tile(coarse, sact + ARR,      row0, tid);
  stage_tile(motif,  sact + 2 * ARR,  row0, tid);
  __syncthreads();

  const int wave = tid >> 5;
  const int lane = tid & 31;
  const int lo   = lane & 15;   // column / row-within-tile selector
  const int hi   = lane >> 4;   // K-half selector
  const int arow = wave * 16 + lo;

  // motif A-fragments are reused by all 8 heads: hoist them once (8 x 8 VGPRs)
  bf16x16 am[HID / 32];
#pragma unroll
  for (int kb = 0; kb < HID / 32; ++kb)
    am[kb] = lds_afrag(sact + 2 * ARR, arow, kb, hi);

  // ---- per-head QKV GEMMs + fused 2-way softmax attention ----
  for (int h = 0; h < HEADS; ++h) {
    f32x8 q0 = vzero8(), q1 = vzero8();
    f32x8 k00 = vzero8(), k01 = vzero8(), k10 = vzero8(), k11 = vzero8();
    f32x8 v00 = vzero8(), v01 = vzero8(), v10 = vzero8(), v11 = vzero8();

#pragma unroll
    for (int kb = 0; kb < HID / 32; ++kb) {
      bf16x16 af = lds_afrag(sact,       arow, kb, hi);  // fine
      bf16x16 ac = lds_afrag(sact + ARR, arow, kb, hi);  // coarse
      {   // N-tile 0: head cols h*32 .. h*32+15
        bf16x16 bq = gbl_bfrag(wq, h * 32 + lo, kb, hi);
        bf16x16 bk = gbl_bfrag(wk, h * 32 + lo, kb, hi);
        bf16x16 bv = gbl_bfrag(wv, h * 32 + lo, kb, hi);
        q0  = WMMA(am[kb], bq, q0);
        k00 = WMMA(af, bk, k00);  k10 = WMMA(ac, bk, k10);
        v00 = WMMA(af, bv, v00);  v10 = WMMA(ac, bv, v10);
      }
      {   // N-tile 1: head cols h*32+16 .. h*32+31
        bf16x16 bq = gbl_bfrag(wq, h * 32 + 16 + lo, kb, hi);
        bf16x16 bk = gbl_bfrag(wk, h * 32 + 16 + lo, kb, hi);
        bf16x16 bv = gbl_bfrag(wv, h * 32 + 16 + lo, kb, hi);
        q1  = WMMA(am[kb], bq, q1);
        k01 = WMMA(af, bk, k01);  k11 = WMMA(ac, bk, k11);
        v01 = WMMA(af, bv, v01);  v11 = WMMA(ac, bv, v11);
      }
    }

    // scores: per row, sum over d_k=32 of q*k  (same C-layout -> elementwise mul,
    // reduce across each 16-lane half: reg r holds row r (lanes 0-15) / row r+8 (lanes 16-31))
    f32x8 p0 = q0 * k00 + q1 * k01;   // s=0 (fine)
    f32x8 p1 = q0 * k10 + q1 * k11;   // s=1 (coarse)
#pragma unroll
    for (int m = 1; m <= 8; m <<= 1) {
#pragma unroll
      for (int r = 0; r < 8; ++r) {
        p0[r] += __shfl_xor(p0[r], m, 32);
        p1[r] += __shfl_xor(p1[r], m, 32);
      }
    }
    // softmax over s (scale by 1/d_k, NOT 1/sqrt(d_k), per reference)
    f32x8 a0, a1;
#pragma unroll
    for (int r = 0; r < 8; ++r) {
      float d  = (p1[r] - p0[r]) * (1.0f / 32.0f);
      float e  = __expf(d);
      float w1 = e / (1.0f + e);
      a1[r] = w1;
      a0[r] = 1.0f - w1;
    }
    f32x8 c0 = a0 * v00 + a1 * v10;
    f32x8 c1 = a0 * v01 + a1 * v11;

    // write head context to per-wave LDS rows as bf16 (C layout -> row-major)
#pragma unroll
    for (int r = 0; r < 8; ++r) {
      int rrow = wave * 16 + r + 8 * hi;
      sctx[rrow * LDST + h * 32 + lo]      = (__bf16)c0[r];
      sctx[rrow * LDST + h * 32 + 16 + lo] = (__bf16)c1[r];
    }
  }

  // per-wave RAW through LDS; be explicit about the DS counter
  asm volatile("s_wait_dscnt 0" ::: "memory");

  // ---- output projection: out = ctx @ Wo^T + bo ----
  // context A-fragments are reused by all 16 N-tiles: hoist them once (head regs are dead now)
  bf16x16 actx[HID / 32];
#pragma unroll
  for (int kb = 0; kb < HID / 32; ++kb)
    actx[kb] = lds_afrag(sctx, arow, kb, hi);

  for (int nt = 0; nt < HID / 16; ++nt) {
    f32x8 o = vzero8();
#pragma unroll
    for (int kb = 0; kb < HID / 32; ++kb) {
      bf16x16 b = gbl_bfrag(wo, nt * 16 + lo, kb, hi);
      o = WMMA(actx[kb], b, o);
    }
    const int   col  = nt * 16 + lo;
    const float bias = bo[col];
#pragma unroll
    for (int r = 0; r < 8; ++r) {
      out[(size_t)(row0 + wave * 16 + r + 8 * hi) * HID + col] = o[r] + bias;
    }
  }
}

// ---------------- launch ----------------
extern "C" void kernel_launch(void* const* d_in, const int* in_sizes, int n_in,
                              void* d_out, int out_size, void* d_ws, size_t ws_size,
                              hipStream_t stream) {
  const float* fine   = (const float*)d_in[0];
  const float* coarse = (const float*)d_in[1];
  const float* motif  = (const float*)d_in[2];
  const float* Wq     = (const float*)d_in[3];
  const float* Wk     = (const float*)d_in[4];
  const float* Wv     = (const float*)d_in[5];
  const float* Wo     = (const float*)d_in[6];
  const float* bo     = (const float*)d_in[7];
  float*       out    = (float*)d_out;

  __bf16* wbf = (__bf16*)d_ws;   // 4 * 256*256 bf16 = 512 KB

  // 1) weights -> bf16 (L2-resident thereafter)
  la_cvt_weights<<<(HID * HID + 255) / 256, 256, 0, stream>>>(Wq, Wk, Wv, Wo, wbf);

  // 2) fused QKV + attention + output projection
  const int rows    = in_sizes[0] / HID;                 // B = 262144
  const int nblocks = rows / ROWS_PER_BLOCK;             // 4096
  const size_t lds_bytes = (size_t)(3 * ARR + ROWS_PER_BLOCK * LDST) * sizeof(__bf16); // 132 KB
  la_fused<<<nblocks, BLOCK_THREADS, lds_bytes, stream>>>(fine, coarse, motif, wbf, bo, out);
}
